// GCN2_48524540510773
// MI455X (gfx1250) — matrix operand order repaired and checked by reading, not compiled, not run on hardware
//
#include <hip/hip_runtime.h>
#include <hip/hip_bf16.h>
#include <cstdint>
#include <cstddef>

// ---------------------------------------------------------------------------
// CDNA5 (gfx1250) GCN pipeline: bf16 WMMA GEMMs + atomic scatter aggregation
// GEMM v2: each wave computes 1 M-tile x 4 N-tiles (A-fragment register reuse)
// ---------------------------------------------------------------------------

typedef __attribute__((ext_vector_type(16))) __bf16 v16bf;
typedef __attribute__((ext_vector_type(8)))  float  v8f;

static __device__ __forceinline__ unsigned short f2bf(float f) {
  unsigned int u = __float_as_uint(f);
  u += 0x7FFFu + ((u >> 16) & 1u);           // round-to-nearest-even
  return (unsigned short)(u >> 16);
}

// ---- activation fp32 -> bf16, padded rows zeroed --------------------------
__global__ void k_x_to_bf16(const float* __restrict__ X, unsigned short* __restrict__ Xb,
                            long long nValid, long long total) {
  long long idx = (long long)blockIdx.x * blockDim.x + threadIdx.x;
  if (idx >= total) return;
  float v = (idx < nValid) ? X[idx] : 0.0f;   // row stride == Cin, so flat copy works
  Xb[idx] = f2bf(v);
}

// ---- weight fp32 [K x Cout] -> bf16 transposed [Cout x K] -----------------
__global__ void k_w_to_bf16T(const float* __restrict__ W, unsigned short* __restrict__ WT,
                             int Cout, int kShift, long long total) {
  long long idx = (long long)blockIdx.x * blockDim.x + threadIdx.x;
  if (idx >= total) return;
  int c = (int)(idx >> kShift);
  int k = (int)(idx & ((1 << kShift) - 1));
  WT[idx] = f2bf(W[(size_t)k * Cout + c]);
}

// ---- WMMA GEMM: G[Mpad x Nc] = Xb[Mpad x K] * WT[Nc x K]^T ----------------
// 8 waves/block as 2(M) x 4(N-group); each wave: 1 M-tile x 4 N-tiles.
// Block tile: 32(M) x 256(N). A fragment loaded once, used by 4 WMMAs.
__global__ __launch_bounds__(256) void k_gemm_bf16_wmma(
    const unsigned short* __restrict__ Xb,
    const unsigned short* __restrict__ WT,
    float* __restrict__ G, int K, int Nc) {
  const int lane  = threadIdx.x & 31;
  const int wave  = threadIdx.x >> 5;         // 0..7
  const int waveM = wave & 1;                 // 2 M-tiles per block
  const int waveN = wave >> 1;                // 4 N-groups per block
  const int m0    = blockIdx.x * 32 + waveM * 16;
  const int n0    = blockIdx.y * 256 + waveN * 64;   // this wave's 64 columns
  if (n0 >= Nc) return;                        // wave-uniform early out (layer 3)

  const int lo = lane & 15;
  const int hi = lane >> 4;                    // 0: lanes 0-15, 1: lanes 16-31

  const unsigned short* __restrict__ xrow = Xb + (size_t)(m0 + lo) * K;
  const unsigned short* __restrict__ wr0  = WT + (size_t)(n0 +  0 + lo) * K;
  const unsigned short* __restrict__ wr1  = WT + (size_t)(n0 + 16 + lo) * K;
  const unsigned short* __restrict__ wr2  = WT + (size_t)(n0 + 32 + lo) * K;
  const unsigned short* __restrict__ wr3  = WT + (size_t)(n0 + 48 + lo) * K;

  v8f acc0 = {}, acc1 = {}, acc2 = {}, acc3 = {};
  for (int k0 = 0; k0 < K; k0 += 32) {
    union { v16bf v; unsigned int u[8]; } a, b0, b1, b2, b3;
    // A fragment (16x32): contiguous k = 16*hi + 2p, 2p+1 per lane-half, same order as B
#pragma unroll
    for (int p = 0; p < 8; ++p) {
      int k = k0 + (hi << 4) + (p << 1);
      a.u[p] = *(const unsigned int*)(xrow + k);
    }
    // B fragments (32x16): VGPR p holds K = 16*hi + 2p, 2p+1 at this column
#pragma unroll
    for (int p = 0; p < 8; ++p) {
      int k = k0 + (hi << 4) + (p << 1);
      b0.u[p] = *(const unsigned int*)(wr0 + k);
      b1.u[p] = *(const unsigned int*)(wr1 + k);
      b2.u[p] = *(const unsigned int*)(wr2 + k);
      b3.u[p] = *(const unsigned int*)(wr3 + k);
    }
    acc0 = __builtin_amdgcn_wmma_f32_16x16x32_bf16(false, a.v, false, b0.v, (short)0, acc0, false, false);
    acc1 = __builtin_amdgcn_wmma_f32_16x16x32_bf16(false, a.v, false, b1.v, (short)0, acc1, false, false);
    acc2 = __builtin_amdgcn_wmma_f32_16x16x32_bf16(false, a.v, false, b2.v, (short)0, acc2, false, false);
    acc3 = __builtin_amdgcn_wmma_f32_16x16x32_bf16(false, a.v, false, b3.v, (short)0, acc3, false, false);
  }
#pragma unroll
  for (int v = 0; v < 8; ++v) {
    int r = m0 + v + (hi << 3);               // C/D layout: M = v + 8*hi
    float* grow = G + (size_t)r * Nc + lo;
    grow[n0 +  0] = acc0[v];
    grow[n0 + 16] = acc1[v];
    grow[n0 + 32] = acc2[v];
    grow[n0 + 48] = acc3[v];
  }
}

// ---- degree / normalization ----------------------------------------------
__global__ void k_fill1(float* __restrict__ p, int n) {
  int i = blockIdx.x * blockDim.x + threadIdx.x;
  if (i < n) p[i] = 1.0f;                     // self-loop contributes 1
}
__global__ void k_add_deg(const int* __restrict__ dst, float* __restrict__ deg, int E) {
  int i = blockIdx.x * blockDim.x + threadIdx.x;
  if (i < E) atomicAdd(&deg[dst[i]], 1.0f);
}
__global__ void k_rsqrt(float* __restrict__ p, int n) {
  int i = blockIdx.x * blockDim.x + threadIdx.x;
  if (i < n) p[i] = rsqrtf(p[i]);
}

// ---- aggregation ----------------------------------------------------------
__global__ void k_init_selfloop(const float* __restrict__ G, const float* __restrict__ dinv,
                                float* __restrict__ O, int cShift, long long total) {
  long long idx = (long long)blockIdx.x * blockDim.x + threadIdx.x;
  if (idx >= total) return;
  int i = (int)(idx >> cShift);
  float di = dinv[i];
  O[idx] = G[idx] * di * di;
}

__global__ void k_scatter(const int* __restrict__ src, const int* __restrict__ dst,
                          const float* __restrict__ G, float* __restrict__ O,
                          const float* __restrict__ dinv, int cShift, long long total) {
  long long idx = (long long)blockIdx.x * blockDim.x + threadIdx.x;
  if (idx >= total) return;
  int e = (int)(idx >> cShift);
  int c = (int)(idx & ((1 << cShift) - 1));
  int s = src[e], d = dst[e];
  float v = G[((size_t)s << cShift) + c] * dinv[s] * dinv[d];
  atomicAdd(&O[((size_t)d << cShift) + c], v);
}

__global__ void k_bias_elu(float* __restrict__ O, const float* __restrict__ b,
                           int cShift, long long total) {
  long long idx = (long long)blockIdx.x * blockDim.x + threadIdx.x;
  if (idx >= total) return;
  float x = O[idx] + b[idx & ((1 << cShift) - 1)];
  O[idx] = (x > 0.0f) ? x : (__expf(x) - 1.0f);
}

__global__ void k_combine(const float* __restrict__ OA, const float* __restrict__ OB,
                          const float* __restrict__ aw, float* __restrict__ H,
                          int cShift, long long total) {
  long long idx = (long long)blockIdx.x * blockDim.x + threadIdx.x;
  if (idx >= total) return;
  int c = (int)(idx & ((1 << cShift) - 1));
  float a0 = aw[2 * c], a1 = aw[2 * c + 1];
  float m = fmaxf(a0, a1);
  float e0 = __expf(a0 - m), e1 = __expf(a1 - m);
  float w0 = e0 / (e0 + e1);
  H[idx] = OA[idx] * w0 + OB[idx] * (1.0f - w0);
}

// ---------------------------------------------------------------------------
// Host side
// ---------------------------------------------------------------------------
static inline int ilog2i(int x) { int s = 0; while ((1 << s) < x) ++s; return s; }
static inline unsigned nblk(long long total, int bs) { return (unsigned)((total + bs - 1) / bs); }

struct LayerBufs {
  float *G_A, *G_B, *O_A, *O_B;
  unsigned short *Xb, *WTa, *WTb;
};

static void run_layer(const float* Hin, int Cin, int Cout,
                      const float* Wa, const float* ba,
                      const float* Wb, const float* bb,
                      const float* aw,
                      const int* s1, const int* d1, int E1, const float* dinv1,
                      const int* s2, const int* d2, int E2, const float* dinv2,
                      const LayerBufs& B, float* Hout,
                      int Nn, int Mpad, hipStream_t stream) {
  const int kSh = ilog2i(Cin), cSh = ilog2i(Cout);
  const long long totX = (long long)Mpad * Cin;
  const long long nVal = (long long)Nn * Cin;
  k_x_to_bf16<<<nblk(totX, 256), 256, 0, stream>>>(Hin, B.Xb, nVal, totX);

  const long long totW = (long long)Cin * Cout;
  k_w_to_bf16T<<<nblk(totW, 256), 256, 0, stream>>>(Wa, B.WTa, Cout, kSh, totW);
  k_w_to_bf16T<<<nblk(totW, 256), 256, 0, stream>>>(Wb, B.WTb, Cout, kSh, totW);

  dim3 g(Mpad / 32, (Cout + 255) / 256);
  k_gemm_bf16_wmma<<<g, 256, 0, stream>>>(B.Xb, B.WTa, B.G_A, Cin, Cout);
  k_gemm_bf16_wmma<<<g, 256, 0, stream>>>(B.Xb, B.WTb, B.G_B, Cin, Cout);

  const long long totO = (long long)Nn << cSh;
  // branch a (edge set 1)
  k_init_selfloop<<<nblk(totO, 256), 256, 0, stream>>>(B.G_A, dinv1, B.O_A, cSh, totO);
  k_scatter<<<nblk((long long)E1 << cSh, 256), 256, 0, stream>>>(s1, d1, B.G_A, B.O_A, dinv1, cSh, (long long)E1 << cSh);
  k_bias_elu<<<nblk(totO, 256), 256, 0, stream>>>(B.O_A, ba, cSh, totO);
  // branch b (edge set 2)
  k_init_selfloop<<<nblk(totO, 256), 256, 0, stream>>>(B.G_B, dinv2, B.O_B, cSh, totO);
  k_scatter<<<nblk((long long)E2 << cSh, 256), 256, 0, stream>>>(s2, d2, B.G_B, B.O_B, dinv2, cSh, (long long)E2 << cSh);
  k_bias_elu<<<nblk(totO, 256), 256, 0, stream>>>(B.O_B, bb, cSh, totO);
  // learned softmax combine
  k_combine<<<nblk(totO, 256), 256, 0, stream>>>(B.O_A, B.O_B, aw, Hout, cSh, totO);
}

extern "C" void kernel_launch(void* const* d_in, const int* in_sizes, int n_in,
                              void* d_out, int out_size, void* d_ws, size_t ws_size,
                              hipStream_t stream) {
  const float* X   = (const float*)d_in[0];
  const int*   ei1 = (const int*)d_in[1];
  const int*   ei2 = (const int*)d_in[2];
  const float* W11 = (const float*)d_in[3];  const float* b11 = (const float*)d_in[4];
  const float* W12 = (const float*)d_in[5];  const float* b12 = (const float*)d_in[6];
  const float* W21 = (const float*)d_in[7];  const float* b21 = (const float*)d_in[8];
  const float* W22 = (const float*)d_in[9];  const float* b22 = (const float*)d_in[10];
  const float* W31 = (const float*)d_in[11]; const float* b31 = (const float*)d_in[12];
  const float* W32 = (const float*)d_in[13]; const float* b32 = (const float*)d_in[14];
  const float* aw1 = (const float*)d_in[15];
  const float* aw2 = (const float*)d_in[16];
  const float* aw3 = (const float*)d_in[17];

  const int Nn   = in_sizes[0] / 512;          // 10000
  const int Mpad = (Nn + 31) & ~31;            // 10016 (multiple of 32 for GEMM v2)
  const int E1   = in_sizes[1] / 2;            // 80000
  const int E2   = in_sizes[2] / 2;            // 160000
  const int* s1 = ei1;        const int* d1 = ei1 + E1;
  const int* s2 = ei2;        const int* d2 = ei2 + E2;

  // workspace carve-up (256B aligned by construction, except dinv2: 128B)
  char* w = (char*)d_ws;
  const size_t szBig = (size_t)Mpad * 1024 * sizeof(float);
  LayerBufs B;
  B.G_A = (float*)(w);
  B.G_B = (float*)(w + szBig);
  B.O_A = (float*)(w + 2 * szBig);
  B.O_B = (float*)(w + 3 * szBig);
  B.Xb  = (unsigned short*)(w + 4 * szBig);
  B.WTa = B.Xb  + (size_t)Mpad * 1024;
  B.WTb = B.WTa + (size_t)1024 * 1024;
  float* dinv1 = (float*)(B.WTb + (size_t)1024 * 1024);
  float* dinv2 = dinv1 + Mpad;
  (void)ws_size; (void)n_in; (void)out_size;

  // ---- symmetric-norm degree terms (once, both edge sets) ----
  k_fill1<<<nblk(Nn, 256), 256, 0, stream>>>(dinv1, Nn);
  k_fill1<<<nblk(Nn, 256), 256, 0, stream>>>(dinv2, Nn);
  k_add_deg<<<nblk(E1, 256), 256, 0, stream>>>(d1, dinv1, E1);
  k_add_deg<<<nblk(E2, 256), 256, 0, stream>>>(d2, dinv2, E2);
  k_rsqrt<<<nblk(Nn, 256), 256, 0, stream>>>(dinv1, Nn);
  k_rsqrt<<<nblk(Nn, 256), 256, 0, stream>>>(dinv2, Nn);

  // ---- layer 1: 512 -> 1024 (combine result into G_A, reused as next input)
  run_layer(X, 512, 1024, W11, b11, W12, b12, aw1,
            s1, d1, E1, dinv1, s2, d2, E2, dinv2,
            B, B.G_A, Nn, Mpad, stream);
  // ---- layer 2: 1024 -> 512
  run_layer(B.G_A, 1024, 512, W21, b21, W22, b22, aw2,
            s1, d1, E1, dinv1, s2, d2, E2, dinv2,
            B, B.G_A, Nn, Mpad, stream);
  // ---- layer 3: 512 -> 128, combine straight into d_out
  run_layer(B.G_A, 512, 128, W31, b31, W32, b32, aw3,
            s1, d1, E1, dinv1, s2, d2, E2, dinv2,
            B, (float*)d_out, Nn, Mpad, stream);
}
